// LanguageModel_37538014167303
// MI455X (gfx1250) — matrix operand order repaired and checked by reading, not compiled, not run on hardware
//
#include <hip/hip_runtime.h>
#include <math.h>

// ---------------- model constants ----------------
#define VOCAB 50257
#define EMBD  1024
#define NHEAD 16
#define NKV   4
#define HSZ   64
#define NLAY  4
#define BSZ   4
#define SEQ   2048
#define FFND  4096
#define RMSEPS 2e-4f
#define ROPE_THETA 100000.0f
#define MROWS (BSZ * SEQ)        // 8192 token rows

typedef __bf16 bf16_t;
typedef __attribute__((ext_vector_type(16))) __bf16 v16bf;
typedef __attribute__((ext_vector_type(8)))  float  v8f;
typedef __attribute__((ext_vector_type(4)))  float  f32x4;
typedef int v4i __attribute__((vector_size(16)));

#ifndef __has_builtin
#define __has_builtin(x) 0
#endif
#if __has_builtin(__builtin_amdgcn_global_load_async_to_lds_b128) && \
    __has_builtin(__builtin_amdgcn_s_wait_asynccnt)
#define USE_ASYNC_LDS 1
#else
#define USE_ASYNC_LDS 0
#endif

// 16-byte global->LDS copy: async DMA (ASYNCcnt) when available, else via VGPRs.
static __device__ __forceinline__ void async_cp16(void* l, const void* g) {
#if USE_ASYNC_LDS
  __builtin_amdgcn_global_load_async_to_lds_b128(
      (__attribute__((address_space(1))) v4i*)g,
      (__attribute__((address_space(3))) v4i*)l, 0, 0);
#else
  *(f32x4*)l = *(const f32x4*)g;
#endif
}
static __device__ __forceinline__ void wait_async_lds() {
#if USE_ASYNC_LDS
  __builtin_amdgcn_s_wait_asynccnt(0);
#endif
}

static __device__ __forceinline__ v8f zero8() {
  v8f z = {0.f, 0.f, 0.f, 0.f, 0.f, 0.f, 0.f, 0.f};
  return z;
}

// ---------------- bf16 GEMM: C[M,N] = A[M,K] * W[N,K]^T (+bias) ----------------
// Block 128x128, K-step 32, 8 waves in 2x4 grid, each wave 64x32 (4x2 WMMA tiles).
// Double-buffered LDS: stage tile k+1 (async) while computing tile k; one barrier/step.
#define GBM 128
#define GBN 128
#define GBK 32
#define GLD 40   // padded LDS row stride (bf16 elems) to break bank conflicts

__global__ __launch_bounds__(256) void gemm_bf16_kernel(
    const bf16_t* __restrict__ A,    // [M,K] row-major bf16
    const bf16_t* __restrict__ W,    // [N,K] row-major bf16
    const float*  __restrict__ bias, // [N] or nullptr
    float* __restrict__ C,           // [M,N] fp32
    int M, int N, int K)
{
  __shared__ bf16_t As[2][GBM * GLD];
  __shared__ bf16_t Ws[2][GBN * GLD];

  const int tid  = threadIdx.x;
  const int wave = tid >> 5;
  const int lane = tid & 31;
  const int hlf  = lane >> 4;
  const int l16  = lane & 15;
  const int wm   = wave >> 2;   // 0..1
  const int wn   = wave & 3;    // 0..3
  const int m0 = blockIdx.y * GBM;
  const int n0 = blockIdx.x * GBN;

  v8f acc[4][2];
#pragma unroll
  for (int i = 0; i < 4; ++i)
#pragma unroll
    for (int j = 0; j < 2; ++j) acc[i][j] = zero8();

  const int lr  = tid >> 2;        // 0..63 : row within 64-row slab
  const int lcs = (tid & 3) * 8;   // 0,8,16,24 : bf16 col segment (16B)

  auto stage = [&](int b, int k0) {
#pragma unroll
    for (int it = 0; it < 2; ++it) {
      const int row = lr + it * 64;
      async_cp16(As[b] + row * GLD + lcs,
                 A + (size_t)(m0 + row) * K + k0 + lcs);
      const int wrow = n0 + row;
      if (wrow < N)   // OOB rows feed only discarded output columns
        async_cp16(Ws[b] + row * GLD + lcs,
                   W + (size_t)wrow * K + k0 + lcs);
    }
  };

  stage(0, 0);
  wait_async_lds();
  __syncthreads();

  int buf = 0;
  for (int k0 = 0; k0 < K; k0 += GBK) {
    if (k0 + GBK < K) stage(buf ^ 1, k0 + GBK);

    // A-fragments: lane holds row M=l16; K chunks {0-7,16-23} (lanes 0-15) / {8-15,24-31}
    v16bf af[4];
#pragma unroll
    for (int i = 0; i < 4; ++i) {
      const bf16_t* base = As[buf] + (wm * 64 + i * 16 + l16) * GLD;
      f32x4* p = (f32x4*)&af[i];
      p[0] = *(const f32x4*)(base + hlf * 8);
      p[1] = *(const f32x4*)(base + 16 + hlf * 8);
    }
    // B-fragments: lane holds col N=l16; K chunks {0-7,16-23} (lanes 0-15) / {8-15,24-31}
    v16bf bfr[2];
#pragma unroll
    for (int j = 0; j < 2; ++j) {
      const bf16_t* base = Ws[buf] + (wn * 32 + j * 16 + l16) * GLD + hlf * 8;
      f32x4* p = (f32x4*)&bfr[j];
      p[0] = *(const f32x4*)(base);
      p[1] = *(const f32x4*)(base + 16);
    }
#pragma unroll
    for (int i = 0; i < 4; ++i)
#pragma unroll
      for (int j = 0; j < 2; ++j)
        acc[i][j] = __builtin_amdgcn_wmma_f32_16x16x32_bf16(
            false, af[i], false, bfr[j], (short)0, acc[i][j], false, false);

    wait_async_lds();
    __syncthreads();
    buf ^= 1;
  }

#pragma unroll
  for (int j = 0; j < 2; ++j) {
    const int col = n0 + wn * 32 + j * 16 + l16;
    if (col < N) {
      const float bv = bias ? bias[col] : 0.f;
#pragma unroll
      for (int i = 0; i < 4; ++i) {
#pragma unroll
        for (int r = 0; r < 8; ++r) {
          const int row = m0 + wm * 64 + i * 16 + r + hlf * 8;
          C[(size_t)row * N + col] = acc[i][j][r] + bv;
        }
      }
    }
  }
}

// ---------------- WMMA flash attention (causal, GQA h%NKV) ----------------
// Grid: (B*NH, SEQ/128), block 256 = 8 waves, each wave owns 16 q-rows.
// Double-buffered K/V tiles (32 keys), one barrier per key tile.
__global__ __launch_bounds__(256) void flash_attn_kernel(
    const bf16_t* __restrict__ Q,   // [B,NH,T,HS]
    const bf16_t* __restrict__ K,   // [B,NKV,T,HS]
    const bf16_t* __restrict__ Vt,  // [B,NKV,HS,T] (pre-transposed)
    bf16_t* __restrict__ O,         // [B,T,NH*HS]
    int T)
{
  __shared__ bf16_t Ks[2][32 * 72];   // key tile [32][HS], padded stride 72
  __shared__ bf16_t Vs[2][64 * 40];   // V^T tile [HS][32], padded stride 40
  __shared__ bf16_t Ps[8][16 * 40];   // per-wave P transpose scratch

  const int tid  = threadIdx.x;
  const int wave = tid >> 5;
  const int lane = tid & 31;
  const int hlf  = lane >> 4;
  const int l16  = lane & 15;

  const int bh  = blockIdx.x;
  const int b   = bh / NHEAD;
  const int h   = bh % NHEAD;
  const int kvh = h % NKV;            // jnp.tile GQA mapping
  const int q0  = blockIdx.y * 128;
  const int qm  = q0 + wave * 16 + l16;

  const bf16_t* Kb = K  + ((size_t)b * NKV + kvh) * (size_t)T * HSZ;
  const bf16_t* Vb = Vt + ((size_t)b * NKV + kvh) * (size_t)HSZ * T;

  auto stageKV = [&](int bufid, int kbase) {
    const int kr = tid >> 3, ksg = (tid & 7) * 8;
    async_cp16(Ks[bufid] + kr * 72 + ksg,
               Kb + (size_t)(kbase + kr) * HSZ + ksg);
    const int vr = tid >> 2, vsg = (tid & 3) * 8;
    async_cp16(Vs[bufid] + vr * 40 + vsg,
               Vb + (size_t)vr * T + kbase + vsg);
  };

  // Q rows as two A-fragments (K = 64 split into 2x32)
  v16bf qf[2];
  {
    const bf16_t* qrow = Q + ((size_t)bh * T + qm) * HSZ;
#pragma unroll
    for (int c = 0; c < 2; ++c) {
      f32x4* p = (f32x4*)&qf[c];
      p[0] = *(const f32x4*)(qrow + c * 32 + hlf * 8);
      p[1] = *(const f32x4*)(qrow + c * 32 + 16 + hlf * 8);
    }
  }

  float mi[8], li[8];
  v8f oacc[4];
#pragma unroll
  for (int r = 0; r < 8; ++r) { mi[r] = -INFINITY; li[r] = 0.f; }
#pragma unroll
  for (int j = 0; j < 4; ++j) oacc[j] = zero8();

  const int ntiles = (q0 + 128) / 32;   // causal: keys up to q0+127
  stageKV(0, 0);
  wait_async_lds();
  __syncthreads();

  int buf = 0;
  for (int kt = 0; kt < ntiles; ++kt) {
    const int kbase = kt * 32;
    if (kt + 1 < ntiles) stageKV(buf ^ 1, kbase + 32);

    // S = Q * K^T for two 16-key groups
    v8f s0 = zero8(), s1 = zero8();
#pragma unroll
    for (int c = 0; c < 2; ++c) {
      v16bf b0, b1;
      {
        const bf16_t* base = Ks[buf] + l16 * 72 + c * 32 + hlf * 8;
        f32x4* p = (f32x4*)&b0;
        p[0] = *(const f32x4*)base; p[1] = *(const f32x4*)(base + 16);
      }
      {
        const bf16_t* base = Ks[buf] + (16 + l16) * 72 + c * 32 + hlf * 8;
        f32x4* p = (f32x4*)&b1;
        p[0] = *(const f32x4*)base; p[1] = *(const f32x4*)(base + 16);
      }
      s0 = __builtin_amdgcn_wmma_f32_16x16x32_bf16(false, qf[c], false, b0, (short)0, s0, false, false);
      s1 = __builtin_amdgcn_wmma_f32_16x16x32_bf16(false, qf[c], false, b1, (short)0, s1, false, false);
    }

    // online softmax in C-fragment layout: row = r + 8*hlf, col key = l16
    float alpha[8];
#pragma unroll
    for (int r = 0; r < 8; ++r) {
      const int qi = q0 + wave * 16 + r + hlf * 8;
      float v0 = s0[r] * 0.125f;   // 1/sqrt(64)
      float v1 = s1[r] * 0.125f;
      if (kbase + l16      > qi) v0 = -INFINITY;
      if (kbase + 16 + l16 > qi) v1 = -INFINITY;
      float mx = fmaxf(v0, v1);
#pragma unroll
      for (int msk = 8; msk >= 1; msk >>= 1) mx = fmaxf(mx, __shfl_xor(mx, msk, 32));
      const float mnew = fmaxf(mi[r], mx);
      float a, p0, p1;
      if (mnew == -INFINITY) { a = 1.f; p0 = 0.f; p1 = 0.f; }
      else {
        a  = __expf(mi[r] - mnew);
        p0 = __expf(v0 - mnew);
        p1 = __expf(v1 - mnew);
      }
      float rs = p0 + p1;
#pragma unroll
      for (int msk = 8; msk >= 1; msk >>= 1) rs += __shfl_xor(rs, msk, 32);
      li[r] = li[r] * a + rs;
      mi[r] = mnew;
      alpha[r] = a;
      bf16_t* prow = &Ps[wave][(r + hlf * 8) * 40];
      prow[l16]      = (bf16_t)p0;
      prow[16 + l16] = (bf16_t)p1;
    }
#pragma unroll
    for (int j = 0; j < 4; ++j)
#pragma unroll
      for (int r = 0; r < 8; ++r) oacc[j][r] *= alpha[r];

    // P as A-fragment (C->A transpose via per-wave LDS)
    v16bf pf;
    {
      const bf16_t* base = &Ps[wave][l16 * 40];
      f32x4* p = (f32x4*)&pf;
      p[0] = *(const f32x4*)(base + hlf * 8);
      p[1] = *(const f32x4*)(base + 16 + hlf * 8);
    }
    // O += P * V
#pragma unroll
    for (int j = 0; j < 4; ++j) {
      v16bf vfr;
      const bf16_t* base = Vs[buf] + (j * 16 + l16) * 40 + hlf * 8;
      f32x4* p = (f32x4*)&vfr;
      p[0] = *(const f32x4*)base; p[1] = *(const f32x4*)(base + 16);
      oacc[j] = __builtin_amdgcn_wmma_f32_16x16x32_bf16(
          false, pf, false, vfr, (short)0, oacc[j], false, false);
    }

    wait_async_lds();
    __syncthreads();
    buf ^= 1;
  }

#pragma unroll
  for (int r = 0; r < 8; ++r) {
    const int t = q0 + wave * 16 + r + hlf * 8;
    const float inv = (li[r] > 0.f) ? (1.f / li[r]) : 0.f;
    bf16_t* orow = O + ((size_t)(b * T + t)) * (NHEAD * HSZ) + h * HSZ;
#pragma unroll
    for (int j = 0; j < 4; ++j)
      orow[j * 16 + l16] = (bf16_t)(oacc[j][r] * inv);
  }
}

// ---------------- elementwise helpers ----------------
static __device__ float block_reduce_sum(float v) {
  __shared__ float red[8];
  __shared__ float tot;
#pragma unroll
  for (int m = 16; m >= 1; m >>= 1) v += __shfl_xor(v, m, 32);
  const int lane = threadIdx.x & 31, w = threadIdx.x >> 5;
  if (lane == 0) red[w] = v;
  __syncthreads();
  if (threadIdx.x == 0) {
    float s = 0.f;
    for (int i = 0; i < 8; ++i) s += red[i];
    tot = s;
  }
  __syncthreads();
  return tot;
}

__global__ __launch_bounds__(256) void embed_kernel(
    const int* __restrict__ tok, const float* __restrict__ emb,
    float* __restrict__ X)
{
  const int row = blockIdx.x;
  const int t = tok[row];
  const float* src = emb + (size_t)t * EMBD;
  float* dst = X + (size_t)row * EMBD;
  for (int i = threadIdx.x; i < EMBD; i += 256) dst[i] = src[i];
}

__global__ __launch_bounds__(256) void rmsnorm_bf16_kernel(
    const float* __restrict__ X, bf16_t* __restrict__ Y, int D)
{
  const float* x = X + (size_t)blockIdx.x * D;
  float ss = 0.f;
  for (int i = threadIdx.x; i < D; i += 256) { float v = x[i]; ss += v * v; }
  ss = block_reduce_sum(ss);
  const float sc = rsqrtf(ss / (float)D + RMSEPS);
  bf16_t* y = Y + (size_t)blockIdx.x * D;
  for (int i = threadIdx.x; i < D; i += 256) y[i] = (bf16_t)(x[i] * sc);
}

__global__ __launch_bounds__(256) void residual_rms_kernel(
    float* __restrict__ X, const float* __restrict__ Yv, int D)
{
  const float* y = Yv + (size_t)blockIdx.x * D;
  float ss = 0.f;
  for (int i = threadIdx.x; i < D; i += 256) { float v = y[i]; ss += v * v; }
  ss = block_reduce_sum(ss);
  const float sc = rsqrtf(ss / (float)D + RMSEPS);
  float* x = X + (size_t)blockIdx.x * D;
  for (int i = threadIdx.x; i < D; i += 256) x[i] += y[i] * sc;
}

__global__ __launch_bounds__(256) void rope_cvt_kernel(
    const float* __restrict__ X,  // [B*T, H*HS] fp32
    bf16_t* __restrict__ Y,       // [B,H,T,HS] bf16
    int H, int n)                 // n = B*T*H*(HS/2)
{
  const int idx = blockIdx.x * 256 + threadIdx.x;
  if (idx >= n) return;
  const int i = idx & 31;
  int tmp = idx >> 5;
  const int h = tmp % H; tmp /= H;
  const int t = tmp % SEQ;
  const int b = tmp / SEQ;
  const float inv = __powf(ROPE_THETA, -((float)(2 * i)) / (float)HSZ);
  float sn, cs;
  __sincosf((float)t * inv, &sn, &cs);
  const float* src = X + ((size_t)(b * SEQ + t)) * ((size_t)H * HSZ) + h * HSZ + 2 * i;
  const float x0 = src[0], x1 = src[1];
  bf16_t* dst = Y + (((size_t)b * H + h) * SEQ + t) * HSZ + 2 * i;
  dst[0] = (bf16_t)(x0 * cs - x1 * sn);
  dst[1] = (bf16_t)(x0 * sn + x1 * cs);
}

__global__ __launch_bounds__(256) void v_trans_kernel(
    const float* __restrict__ V,  // [B*T, NKV*HS] fp32
    bf16_t* __restrict__ Vt,      // [B,NKV,HS,T] bf16
    int n)                        // n = B*T*NKV*HS
{
  const int idx = blockIdx.x * 256 + threadIdx.x;
  if (idx >= n) return;
  const int hs = idx & 63;
  int tmp = idx >> 6;
  const int h = tmp % NKV; tmp /= NKV;
  const int t = tmp % SEQ;
  const int b = tmp / SEQ;
  Vt[(((size_t)b * NKV + h) * HSZ + hs) * SEQ + t] =
      (bf16_t)V[((size_t)(b * SEQ + t)) * (NKV * HSZ) + h * HSZ + hs];
}

__global__ __launch_bounds__(256) void gelu_mul_kernel(
    const float* __restrict__ H1, const float* __restrict__ H3,
    bf16_t* __restrict__ G, size_t n)
{
  const size_t idx = (size_t)blockIdx.x * 256 + threadIdx.x;
  if (idx >= n) return;
  const float x = H1[idx];
  const float t = tanhf(0.7978845608028654f * (x + 0.044715f * x * x * x));
  G[idx] = (bf16_t)(0.5f * x * (1.f + t) * H3[idx]);
}

__global__ __launch_bounds__(256) void cvt_bf16_kernel(
    const float* __restrict__ S, bf16_t* __restrict__ D, size_t n)
{
  const size_t idx = (size_t)blockIdx.x * 256 + threadIdx.x;
  if (idx < n) D[idx] = (bf16_t)S[idx];
}

// ---------------- host driver ----------------
extern "C" void kernel_launch(void* const* d_in, const int* in_sizes, int n_in,
                              void* d_out, int out_size, void* d_ws, size_t ws_size,
                              hipStream_t stream)
{
  (void)in_sizes; (void)n_in; (void)out_size; (void)ws_size;
  const int*   tokens = (const int*)  d_in[0];
  const float* emb    = (const float*)d_in[1];
  const float* lm_w   = (const float*)d_in[2 + 10 * NLAY];
  const float* lm_b   = (const float*)d_in[3 + 10 * NLAY];

  char* ws = (char*)d_ws;
  size_t off = 0;
  auto carve = [&](size_t bytes) -> char* {
    char* p = ws + off;
    off += (bytes + 255) & ~(size_t)255;
    return p;
  };

  bf16_t *wq[NLAY], *wk[NLAY], *wv[NLAY], *wo[NLAY], *w1[NLAY], *w2[NLAY], *w3[NLAY];
  for (int l = 0; l < NLAY; ++l) {
    wq[l] = (bf16_t*)carve((size_t)EMBD * EMBD * 2);
    wk[l] = (bf16_t*)carve((size_t)NKV * HSZ * EMBD * 2);
    wv[l] = (bf16_t*)carve((size_t)NKV * HSZ * EMBD * 2);
    wo[l] = (bf16_t*)carve((size_t)EMBD * EMBD * 2);
    w1[l] = (bf16_t*)carve((size_t)FFND * EMBD * 2);
    w2[l] = (bf16_t*)carve((size_t)EMBD * FFND * 2);
    w3[l] = (bf16_t*)carve((size_t)FFND * EMBD * 2);
  }
  bf16_t* lmw_bf = (bf16_t*)carve((size_t)VOCAB * EMBD * 2);

  float*  x    = (float*) carve((size_t)MROWS * EMBD * 4);   // residual stream fp32
  bf16_t* xn   = (bf16_t*)carve((size_t)MROWS * EMBD * 2);   // normalized bf16
  float*  bigA = (float*) carve((size_t)MROWS * FFND * 4);   // q / o / h1 / ffn-out
  float*  bigB = (float*) carve((size_t)MROWS * FFND * 4);   // k,v / h3
  bf16_t* qbf  = (bf16_t*)carve((size_t)BSZ * NHEAD * SEQ * HSZ * 2);
  bf16_t* kbf  = (bf16_t*)carve((size_t)BSZ * NKV * SEQ * HSZ * 2);
  bf16_t* vtbf = (bf16_t*)carve((size_t)BSZ * NKV * HSZ * SEQ * 2);
  bf16_t* abf  = (bf16_t*)carve((size_t)MROWS * EMBD * 2);
  bf16_t* gbf  = (bf16_t*)carve((size_t)MROWS * FFND * 2);

  auto cvt = [&](const float* s, bf16_t* d, size_t n) {
    cvt_bf16_kernel<<<(int)((n + 255) / 256), 256, 0, stream>>>(s, d, n);
  };
  for (int l = 0; l < NLAY; ++l) {
    const int base = 2 + 10 * l;
    cvt((const float*)d_in[base + 0], wq[l], (size_t)EMBD * EMBD);
    cvt((const float*)d_in[base + 1], wk[l], (size_t)NKV * HSZ * EMBD);
    cvt((const float*)d_in[base + 2], wv[l], (size_t)NKV * HSZ * EMBD);
    cvt((const float*)d_in[base + 3], wo[l], (size_t)EMBD * EMBD);
    cvt((const float*)d_in[base + 4], w1[l], (size_t)FFND * EMBD);
    cvt((const float*)d_in[base + 6], w2[l], (size_t)EMBD * FFND);
    cvt((const float*)d_in[base + 8], w3[l], (size_t)FFND * EMBD);
  }
  cvt(lm_w, lmw_bf, (size_t)VOCAB * EMBD);

  auto gemm = [&](const bf16_t* A, const bf16_t* W, const float* bias,
                  float* C, int M, int N, int K) {
    dim3 grid((N + GBN - 1) / GBN, (M + GBM - 1) / GBM);
    gemm_bf16_kernel<<<grid, 256, 0, stream>>>(A, W, bias, C, M, N, K);
  };

  embed_kernel<<<MROWS, 256, 0, stream>>>(tokens, emb, x);

  for (int l = 0; l < NLAY; ++l) {
    const int base = 2 + 10 * l;
    const float* b1 = (const float*)d_in[base + 5];
    const float* b2 = (const float*)d_in[base + 7];
    const float* b3 = (const float*)d_in[base + 9];

    rmsnorm_bf16_kernel<<<MROWS, 256, 0, stream>>>(x, xn, EMBD);

    float* qf32 = bigA;
    float* kf32 = bigB;
    float* vf32 = bigB + (size_t)MROWS * NKV * HSZ;
    gemm(xn, wq[l], nullptr, qf32, MROWS, EMBD, EMBD);
    gemm(xn, wk[l], nullptr, kf32, MROWS, NKV * HSZ, EMBD);
    gemm(xn, wv[l], nullptr, vf32, MROWS, NKV * HSZ, EMBD);

    {
      int n = BSZ * SEQ * NHEAD * (HSZ / 2);
      rope_cvt_kernel<<<(n + 255) / 256, 256, 0, stream>>>(qf32, qbf, NHEAD, n);
      n = BSZ * SEQ * NKV * (HSZ / 2);
      rope_cvt_kernel<<<(n + 255) / 256, 256, 0, stream>>>(kf32, kbf, NKV, n);
      n = BSZ * SEQ * NKV * HSZ;
      v_trans_kernel<<<(n + 255) / 256, 256, 0, stream>>>(vf32, vtbf, n);
    }

    flash_attn_kernel<<<dim3(BSZ * NHEAD, SEQ / 128), 256, 0, stream>>>(
        qbf, kbf, vtbf, abf, SEQ);

    float* of32 = bigA;  // q no longer needed
    gemm(abf, wo[l], nullptr, of32, MROWS, EMBD, EMBD);
    residual_rms_kernel<<<MROWS, 256, 0, stream>>>(x, of32, EMBD);

    rmsnorm_bf16_kernel<<<MROWS, 256, 0, stream>>>(x, xn, EMBD);
    float* h1 = bigA;
    float* h3 = bigB;
    gemm(xn, w1[l], b1, h1, MROWS, FFND, EMBD);
    gemm(xn, w3[l], b3, h3, MROWS, FFND, EMBD);
    {
      size_t n = (size_t)MROWS * FFND;
      gelu_mul_kernel<<<(int)((n + 255) / 256), 256, 0, stream>>>(h1, h3, gbf, n);
    }
    float* ffo = bigA;   // h1 no longer needed
    gemm(gbf, w2[l], b2, ffo, MROWS, EMBD, FFND);
    residual_rms_kernel<<<MROWS, 256, 0, stream>>>(x, ffo, EMBD);
  }

  rmsnorm_bf16_kernel<<<MROWS, 256, 0, stream>>>(x, xn, EMBD);
  gemm(xn, lmw_bf, lm_b, (float*)d_out, MROWS, VOCAB, EMBD);
}
